// TriModal_42803644072277
// MI455X (gfx1250) — compile-verified, matrix operand order checked by reading, not run on hardware
//
#include <hip/hip_runtime.h>
#include <cstdint>
#include <cstddef>

// Problem constants (match reference)
#define Bn   16
#define Nn   512
#define Dn   512
#define Hn   8
#define DKn  64
#define DFFn 2048
#define GHn  128
#define HCn  4096
#define Mrows (Bn * Nn)   // 8192

typedef __attribute__((ext_vector_type(16))) __bf16 v16bf;
typedef __attribute__((ext_vector_type(8)))  float  v8f;

union FragU { uint4 u[2]; v16bf v; unsigned short s[16]; };

__device__ __forceinline__ unsigned short f2bf(float f) {
    unsigned int u = __builtin_bit_cast(unsigned int, f);
    u += 0x7FFFu + ((u >> 16) & 1u);     // round-to-nearest-even
    return (unsigned short)(u >> 16);
}

// NT-operand fragment load for V_WMMA_F32_16X16X32_BF16.
// ISA layout (CDNA5 ISA, 16-bit A 16x32): lane half h=lane>>4, r=lane&15:
//   elems 0..7  = row r, K = k0 + 8h + [0..7]
//   elems 8..15 = row r, K = k0 + 16 + 8h + [0..7]
// B operand (stored as [N,K] row-major, i.e. B^T) packs identically with N playing M.
__device__ __forceinline__ v16bf load_frag_nt(const unsigned short* base, int ld,
                                              int row0, int k0, int lane) {
    const int half = lane >> 4, r = lane & 15;
    const unsigned short* p = base + (size_t)(row0 + r) * ld + k0 + 8 * half;
    FragU f;
    f.u[0] = *reinterpret_cast<const uint4*>(p);
    f.u[1] = *reinterpret_cast<const uint4*>(p + 16);
    return f.v;
}

__device__ __forceinline__ v8f wmma_bf16(v16bf a, v16bf b, v8f c) {
    return __builtin_amdgcn_wmma_f32_16x16x32_bf16(false, a, false, b, (short)0, c, false, false);
}

// ---------------- elementwise helpers ----------------

__global__ void k_cvt(const float* __restrict__ s, unsigned short* __restrict__ d, size_t n) {
    size_t i = (size_t)blockIdx.x * blockDim.x + threadIdx.x;
    if (i < n) d[i] = f2bf(s[i]);
}

__global__ void k_bias_cvt(const float* __restrict__ s, const float* __restrict__ bias,
                           unsigned short* __restrict__ d, int cols, size_t n) {
    size_t i = (size_t)blockIdx.x * blockDim.x + threadIdx.x;
    if (i < n) {
        int c = (int)(i % (size_t)cols);
        d[i] = f2bf(s[i] + bias[c]);
    }
}

__global__ void k_bias_gelu_cvt(const float* __restrict__ s, const float* __restrict__ bias,
                                unsigned short* __restrict__ d, int cols, size_t n) {
    size_t i = (size_t)blockIdx.x * blockDim.x + threadIdx.x;
    if (i < n) {
        int c = (int)(i % (size_t)cols);
        float x = s[i] + bias[c];
        float g = 0.5f * x * (1.0f + erff(x * 0.70710678118654752f));  // exact GELU
        d[i] = f2bf(g);
    }
}

// ---------------- tiled transpose (+optional bias over src cols), f32 -> bf16 ----------------
// src: [batch][R][C] f32; dst: [batch][C][R] bf16; dst[b][c][r] = src[b][r][c] + bias[c]
__global__ __launch_bounds__(256) void k_transpose_cvt(const float* __restrict__ src,
                                                       const float* __restrict__ bias,
                                                       unsigned short* __restrict__ dst,
                                                       int R, int C) {
    __shared__ float tile[32][33];
    const int b  = blockIdx.z;
    const int r0 = blockIdx.y * 32, c0 = blockIdx.x * 32;
    const int tx = threadIdx.x, ty = threadIdx.y;  // (32, 8)
    const float* sp = src + (size_t)b * R * C;
    unsigned short* dp = dst + (size_t)b * R * C;
    float badd = bias ? bias[c0 + tx] : 0.0f;
#pragma unroll
    for (int i = 0; i < 4; ++i) {
        int r = ty + 8 * i;
        tile[r][tx] = sp[(size_t)(r0 + r) * C + c0 + tx] + badd;
    }
    __syncthreads();
#pragma unroll
    for (int i = 0; i < 4; ++i) {
        int rr = ty + 8 * i;  // output-row (c index) within tile
        dp[(size_t)(c0 + rr) * R + r0 + tx] = f2bf(tile[tx][rr]);
    }
}

// ---------------- WMMA NT GEMM: C[M,N] f32 = A[M,K]bf16 * B[N,K]bf16^T ----------------
// Block: 256 threads = 8 waves as 4(M) x 2(N); macro tile 128x128; wave tile 32x64.
// 8 WMMAs per 6 fragment loads per k-step (~21 FLOP/byte operand intensity).
// Requires M%128==0, N%128==0, K%32==0 (true for every call here).
__global__ __launch_bounds__(256) void k_gemm_nt(const unsigned short* __restrict__ A,
                                                 const unsigned short* __restrict__ Bm,
                                                 float* __restrict__ C,
                                                 int M, int N, int K) {
    const int lane = threadIdx.x & 31;
    const int wave = threadIdx.x >> 5;
    const int m0 = blockIdx.y * 128 + (wave & 3) * 32;
    const int n0 = blockIdx.x * 128 + (wave >> 2) * 64;
    v8f acc[2][4] = {};
    for (int k0 = 0; k0 < K; k0 += 32) {
        v16bf a0 = load_frag_nt(A,  K, m0,      k0, lane);
        v16bf a1 = load_frag_nt(A,  K, m0 + 16, k0, lane);
        v16bf b0 = load_frag_nt(Bm, K, n0,      k0, lane);
        v16bf b1 = load_frag_nt(Bm, K, n0 + 16, k0, lane);
        v16bf b2 = load_frag_nt(Bm, K, n0 + 32, k0, lane);
        v16bf b3 = load_frag_nt(Bm, K, n0 + 48, k0, lane);
        acc[0][0] = wmma_bf16(a0, b0, acc[0][0]);
        acc[0][1] = wmma_bf16(a0, b1, acc[0][1]);
        acc[0][2] = wmma_bf16(a0, b2, acc[0][2]);
        acc[0][3] = wmma_bf16(a0, b3, acc[0][3]);
        acc[1][0] = wmma_bf16(a1, b0, acc[1][0]);
        acc[1][1] = wmma_bf16(a1, b1, acc[1][1]);
        acc[1][2] = wmma_bf16(a1, b2, acc[1][2]);
        acc[1][3] = wmma_bf16(a1, b3, acc[1][3]);
    }
    const int cr = (lane >> 4) * 8, cc = lane & 15;
#pragma unroll
    for (int i = 0; i < 2; ++i)
#pragma unroll
        for (int j = 0; j < 4; ++j) {
            float* cp = C + (size_t)(m0 + i * 16 + cr) * N + n0 + j * 16 + cc;
#pragma unroll
            for (int r = 0; r < 8; ++r) cp[(size_t)r * N] = acc[i][j][r];
        }
}

// ---------------- fused attention: one wave per (b, h, 16-query tile) ----------------
// qs, ks: [B,N,D] bf16 (head h in cols h*64..). vt: [B,D,N] bf16 (V transposed).
// out obf: [B,N,D] bf16 (heads concatenated, ready for Wo GEMM).
#define SLD 516
__global__ __launch_bounds__(32) void k_attn(const unsigned short* __restrict__ qs,
                                             const unsigned short* __restrict__ ks,
                                             const unsigned short* __restrict__ vt,
                                             unsigned short* __restrict__ obf) {
    __shared__ float sc[16 * SLD];
    const int lane = threadIdx.x;
    const int tile = blockIdx.x;             // B*H*(N/16) = 4096
    const int q0 = (tile & 31) * 16;
    const int h  = (tile >> 5) & 7;
    const int b  = tile >> 8;
    const unsigned short* qb = qs + (size_t)b * Nn * Dn;
    const unsigned short* kb = ks + (size_t)b * Nn * Dn;
    const unsigned short* vb = vt + (size_t)b * Dn * Nn;

    const v16bf qf0 = load_frag_nt(qb, Dn, q0, h * DKn,      lane);
    const v16bf qf1 = load_frag_nt(qb, Dn, q0, h * DKn + 32, lane);

    const int cr = (lane >> 4) * 8, cc = lane & 15;
    // Pass 1: scores -> LDS
    for (int n0 = 0; n0 < Nn; n0 += 16) {
        v16bf kf0 = load_frag_nt(kb, Dn, n0, h * DKn,      lane);
        v16bf kf1 = load_frag_nt(kb, Dn, n0, h * DKn + 32, lane);
        v8f sacc = {};
        sacc = wmma_bf16(qf0, kf0, sacc);
        sacc = wmma_bf16(qf1, kf1, sacc);
#pragma unroll
        for (int r = 0; r < 8; ++r) sc[(cr + r) * SLD + n0 + cc] = sacc[r] * 0.125f;
    }
    __syncthreads();

    // Row softmax stats: lane handles row (lane&15), half (lane>>4) of the columns.
    const int row = lane & 15, part = lane >> 4;
    float mx = -1e30f;
    for (int c = part * 256; c < part * 256 + 256; ++c) mx = fmaxf(mx, sc[row * SLD + c]);
    mx = fmaxf(mx, __shfl_xor(mx, 16, 32));
    float sm = 0.0f;
    for (int c = part * 256; c < part * 256 + 256; ++c) sm += __expf(sc[row * SLD + c] - mx);
    sm += __shfl_xor(sm, 16, 32);
    const float inv = 1.0f / sm;

    // Pass 2: O += P * V  (P rebuilt in A-fragment layout directly from LDS)
    v8f oacc[4] = {};
    for (int n0 = 0; n0 < Nn; n0 += 32) {
        FragU pf;
#pragma unroll
        for (int j = 0; j < 8; ++j) {
            pf.s[j]     = f2bf(__expf(sc[row * SLD + n0 + 8 * part + j]      - mx) * inv);
            pf.s[j + 8] = f2bf(__expf(sc[row * SLD + n0 + 16 + 8 * part + j] - mx) * inv);
        }
#pragma unroll
        for (int d = 0; d < 4; ++d) {
            v16bf vf = load_frag_nt(vb, Nn, h * DKn + d * 16, n0, lane);
            oacc[d] = wmma_bf16(pf.v, vf, oacc[d]);
        }
    }
#pragma unroll
    for (int d = 0; d < 4; ++d)
#pragma unroll
        for (int r = 0; r < 8; ++r)
            obf[((size_t)b * Nn + q0 + cr + r) * Dn + h * DKn + d * 16 + cc] = f2bf(oacc[d][r]);
}

// ---------------- residual + bias + LayerNorm over D=512 (one block per row) ----------------
__global__ __launch_bounds__(256) void k_add_ln(const float* __restrict__ resid,
                                                const float* __restrict__ proj,
                                                const float* __restrict__ bias,
                                                const float* __restrict__ gam,
                                                const float* __restrict__ bet,
                                                float* __restrict__ outf,
                                                unsigned short* __restrict__ outbf) {
    __shared__ float rs[256], rq[256];
    const int row = blockIdx.x, t = threadIdx.x;
    const size_t base = (size_t)row * Dn;
    float x0 = resid[base + t]       + proj[base + t]       + bias[t];
    float x1 = resid[base + t + 256] + proj[base + t + 256] + bias[t + 256];
    rs[t] = x0 + x1;
    rq[t] = x0 * x0 + x1 * x1;
    __syncthreads();
    for (int o = 128; o > 0; o >>= 1) {
        if (t < o) { rs[t] += rs[t + o]; rq[t] += rq[t + o]; }
        __syncthreads();
    }
    const float mean = rs[0] * (1.0f / 512.0f);
    const float var  = rq[0] * (1.0f / 512.0f) - mean * mean;
    const float ivs  = rsqrtf(var + 1e-5f);
    float y0 = (x0 - mean) * ivs * gam[t]       + bet[t];
    float y1 = (x1 - mean) * ivs * gam[t + 256] + bet[t + 256];
    if (outf)  { outf[base + t] = y0;          outf[base + t + 256] = y1; }
    if (outbf) { outbf[base + t] = f2bf(y0);   outbf[base + t + 256] = f2bf(y1); }
}

// ---------------- gate head: LN(128) -> ReLU -> Linear(128,8) -> softmax(8); 1 wave/row ----
__global__ __launch_bounds__(256) void k_gate(const float* __restrict__ g1,
                                              const float* __restrict__ bg1,
                                              const float* __restrict__ lg,
                                              const float* __restrict__ lb,
                                              const float* __restrict__ Wg2,
                                              const float* __restrict__ bg2,
                                              float* __restrict__ gates, int rows) {
    const int lane = threadIdx.x & 31;
    const int row  = blockIdx.x * 8 + (threadIdx.x >> 5);
    if (row >= rows) return;
    float h[4];
    float lsum = 0.0f, lsq = 0.0f;
#pragma unroll
    for (int i = 0; i < 4; ++i) {
        float v = g1[(size_t)row * GHn + lane * 4 + i] + bg1[lane * 4 + i];
        h[i] = v; lsum += v; lsq += v * v;
    }
    for (int m = 16; m > 0; m >>= 1) { lsum += __shfl_xor(lsum, m, 32); lsq += __shfl_xor(lsq, m, 32); }
    const float mean = lsum * (1.0f / 128.0f);
    const float var  = lsq  * (1.0f / 128.0f) - mean * mean;
    const float ivs  = rsqrtf(var + 1e-5f);
#pragma unroll
    for (int i = 0; i < 4; ++i) {
        float y = (h[i] - mean) * ivs * lg[lane * 4 + i] + lb[lane * 4 + i];
        h[i] = fmaxf(y, 0.0f);
    }
    float logit[8];
#pragma unroll
    for (int o = 0; o < 8; ++o) {
        float p = 0.0f;
#pragma unroll
        for (int i = 0; i < 4; ++i) p += h[i] * Wg2[o * GHn + lane * 4 + i];
        for (int m = 16; m > 0; m >>= 1) p += __shfl_xor(p, m, 32);
        logit[o] = p + bg2[o];
    }
    if (lane == 0) {
        float mx = logit[0];
#pragma unroll
        for (int o = 1; o < 8; ++o) mx = fmaxf(mx, logit[o]);
        float s = 0.0f;
#pragma unroll
        for (int o = 0; o < 8; ++o) { logit[o] = __expf(logit[o] - mx); s += logit[o]; }
        const float is = 1.0f / s;
#pragma unroll
        for (int o = 0; o < 8; ++o) gates[(size_t)row * 8 + o] = logit[o] * is;
    }
}

// ---------------- fused[b,c,n] = sum_h cma[b, h*C + c, n] * gates[b,n,h] ----------------
__global__ void k_fuse(const float* __restrict__ cma, const float* __restrict__ gates,
                       float* __restrict__ fused) {
    size_t idx = (size_t)blockIdx.x * blockDim.x + threadIdx.x;
    const size_t total = (size_t)Bn * 512 * Nn;
    if (idx >= total) return;
    const int n = (int)(idx & 511);
    const int c = (int)((idx >> 9) & 511);
    const int b = (int)(idx >> 18);
    const float* gp = gates + ((size_t)b * Nn + n) * Hn;
    const float* cp = cma + ((size_t)b * HCn + c) * Nn + n;
    float acc = 0.0f;
#pragma unroll
    for (int hh = 0; hh < Hn; ++hh) acc += cp[(size_t)hh * 512 * Nn] * gp[hh];
    fused[idx] = acc;
}

// =====================================================================================

extern "C" void kernel_launch(void* const* d_in, const int* in_sizes, int n_in,
                              void* d_out, int out_size, void* d_ws, size_t ws_size,
                              hipStream_t stream) {
    (void)in_sizes; (void)n_in; (void)out_size; (void)ws_size;
    const float* q     = (const float*)d_in[0];
    const float* k     = (const float*)d_in[1];
    const float* v     = (const float*)d_in[2];
    const float* cma   = (const float*)d_in[3];
    const float* Wq    = (const float*)d_in[4];
    const float* bq    = (const float*)d_in[5];
    const float* Wk    = (const float*)d_in[6];
    const float* bk    = (const float*)d_in[7];
    const float* Wv    = (const float*)d_in[8];
    const float* bv    = (const float*)d_in[9];
    const float* Wo    = (const float*)d_in[10];
    const float* bo    = (const float*)d_in[11];
    const float* ln1_g = (const float*)d_in[12];
    const float* ln1_b = (const float*)d_in[13];
    const float* ln2_g = (const float*)d_in[14];
    const float* ln2_b = (const float*)d_in[15];
    const float* W1    = (const float*)d_in[16];
    const float* b1    = (const float*)d_in[17];
    const float* W2    = (const float*)d_in[18];
    const float* b2    = (const float*)d_in[19];
    const float* Wg1   = (const float*)d_in[20];
    const float* bg1   = (const float*)d_in[21];
    const float* lng_g = (const float*)d_in[22];
    const float* lng_b = (const float*)d_in[23];
    const float* Wg2   = (const float*)d_in[24];
    const float* bg2   = (const float*)d_in[25];

    // ---- workspace bump allocator ----
    char* w = (char*)d_ws;
    size_t off = 0;
    auto alloc = [&](size_t bytes) -> void* {
        void* p = w + off;
        off = (off + bytes + 255) & ~(size_t)255;
        return p;
    };
    typedef unsigned short u16;
    u16* wbf_q  = (u16*)alloc((size_t)Dn * Dn * 2);
    u16* wbf_k  = (u16*)alloc((size_t)Dn * Dn * 2);
    u16* wbf_v  = (u16*)alloc((size_t)Dn * Dn * 2);
    u16* wbf_o  = (u16*)alloc((size_t)Dn * Dn * 2);
    u16* wbf_1  = (u16*)alloc((size_t)DFFn * Dn * 2);
    u16* wbf_2  = (u16*)alloc((size_t)Dn * DFFn * 2);
    u16* wbf_g1 = (u16*)alloc((size_t)GHn * HCn * 2);
    u16* actA   = (u16*)alloc((size_t)Mrows * Dn * 2);   // bf16 input acts, later o_bf
    u16* qs_bf  = (u16*)alloc((size_t)Mrows * Dn * 2);
    u16* ks_bf  = (u16*)alloc((size_t)Mrows * Dn * 2);
    u16* vt_bf  = (u16*)alloc((size_t)Mrows * Dn * 2);   // [B,D,N]
    float* tmpF = (float*)alloc((size_t)Mrows * Dn * 4); // reused f32 GEMM output
    float* xf   = (float*)alloc((size_t)Mrows * Dn * 4);
    u16* xbf    = (u16*)alloc((size_t)Mrows * Dn * 2);
    float* F1   = (float*)alloc((size_t)Mrows * DFFn * 4);
    u16* F1B    = (u16*)alloc((size_t)Mrows * DFFn * 2);
    float* gatesF = (float*)alloc((size_t)Mrows * Hn * 4);
    u16* cp_bf  = (u16*)F1;  // alias: M*HC*2 bytes == M*DFF*4 bytes; F1 dead by then

    const size_t nMD  = (size_t)Mrows * Dn;
    const size_t nMF  = (size_t)Mrows * DFFn;
    auto eb = [](size_t n) { return (unsigned)((n + 255) / 256); };

    // ---- weights -> bf16 ----
    k_cvt<<<eb((size_t)Dn * Dn), 256, 0, stream>>>(Wq, wbf_q, (size_t)Dn * Dn);
    k_cvt<<<eb((size_t)Dn * Dn), 256, 0, stream>>>(Wk, wbf_k, (size_t)Dn * Dn);
    k_cvt<<<eb((size_t)Dn * Dn), 256, 0, stream>>>(Wv, wbf_v, (size_t)Dn * Dn);
    k_cvt<<<eb((size_t)Dn * Dn), 256, 0, stream>>>(Wo, wbf_o, (size_t)Dn * Dn);
    k_cvt<<<eb((size_t)DFFn * Dn), 256, 0, stream>>>(W1, wbf_1, (size_t)DFFn * Dn);
    k_cvt<<<eb((size_t)Dn * DFFn), 256, 0, stream>>>(W2, wbf_2, (size_t)Dn * DFFn);
    k_cvt<<<eb((size_t)GHn * HCn), 256, 0, stream>>>(Wg1, wbf_g1, (size_t)GHn * HCn);

    const dim3 g_gemm_D(Dn / 128, Mrows / 128);

    // ---- Q projection ----
    k_cvt<<<eb(nMD), 256, 0, stream>>>(q, actA, nMD);
    k_gemm_nt<<<g_gemm_D, 256, 0, stream>>>(actA, wbf_q, tmpF, Mrows, Dn, Dn);
    k_bias_cvt<<<eb(nMD), 256, 0, stream>>>(tmpF, bq, qs_bf, Dn, nMD);
    // ---- K projection ----
    k_cvt<<<eb(nMD), 256, 0, stream>>>(k, actA, nMD);
    k_gemm_nt<<<g_gemm_D, 256, 0, stream>>>(actA, wbf_k, tmpF, Mrows, Dn, Dn);
    k_bias_cvt<<<eb(nMD), 256, 0, stream>>>(tmpF, bk, ks_bf, Dn, nMD);
    // ---- V projection + transpose to [B,D,N] ----
    k_cvt<<<eb(nMD), 256, 0, stream>>>(v, actA, nMD);
    k_gemm_nt<<<g_gemm_D, 256, 0, stream>>>(actA, wbf_v, tmpF, Mrows, Dn, Dn);
    k_transpose_cvt<<<dim3(Dn / 32, Nn / 32, Bn), dim3(32, 8), 0, stream>>>(tmpF, bv, vt_bf, Nn, Dn);

    // ---- fused multi-head attention -> o_bf (reuse actA) ----
    k_attn<<<Bn * Hn * (Nn / 16), 32, 0, stream>>>(qs_bf, ks_bf, vt_bf, actA);

    // ---- output projection + residual + LN1 ----
    k_gemm_nt<<<g_gemm_D, 256, 0, stream>>>(actA, wbf_o, tmpF, Mrows, Dn, Dn);
    k_add_ln<<<Mrows, 256, 0, stream>>>(q, tmpF, bo, ln1_g, ln1_b, xf, xbf);

    // ---- FFN ----
    k_gemm_nt<<<dim3(DFFn / 128, Mrows / 128), 256, 0, stream>>>(xbf, wbf_1, F1, Mrows, DFFn, Dn);
    k_bias_gelu_cvt<<<eb(nMF), 256, 0, stream>>>(F1, b1, F1B, DFFn, nMF);
    k_gemm_nt<<<g_gemm_D, 256, 0, stream>>>(F1B, wbf_2, tmpF, Mrows, Dn, DFFn);
    k_add_ln<<<Mrows, 256, 0, stream>>>(xf, tmpF, b2, ln2_g, ln2_b, (float*)d_out, (u16*)nullptr);

    // ---- gating path: cma^T -> Linear -> LN -> ReLU -> Linear -> softmax ----
    k_transpose_cvt<<<dim3(Nn / 32, HCn / 32, Bn), dim3(32, 8), 0, stream>>>(cma, (const float*)nullptr, cp_bf, HCn, Nn);
    k_gemm_nt<<<dim3(GHn / 128, Mrows / 128), 256, 0, stream>>>(cp_bf, wbf_g1, tmpF, Mrows, GHn, HCn);
    k_gate<<<Mrows / 8, 256, 0, stream>>>(tmpF, bg1, lng_g, lng_b, Wg2, bg2, gatesF, Mrows);
    k_fuse<<<eb((size_t)Bn * 512 * Nn), 256, 0, stream>>>(cma, gatesF, (float*)d_out + (size_t)Bn * Nn * Dn);
}
